// S4Model_83382495085191
// MI455X (gfx1250) — matrix operand order repaired and checked by reading, not compiled, not run on hardware
//
#include <hip/hip_runtime.h>
#include <math.h>

#define B    16
#define LQ   4096
#define DIN  32
#define H    256
#define NC   32
#define NL   4
#define DOUT 10
#define CHUNK 64
#define NCH  (LQ/CHUNK)   // 64
#define LN_EPS 1e-5f

typedef __attribute__((ext_vector_type(16))) __bf16 v16bf;
typedef __attribute__((ext_vector_type(8)))  float  v8f;

__device__ __forceinline__ unsigned short f2bf(float f) {
    unsigned int u = __float_as_uint(f);
    unsigned int r = u + 0x7fffu + ((u >> 16) & 1u);
    return (unsigned short)(r >> 16);
}

// ---------------- encoder: u[b,h,l] = x[b,l,:] . encW[:,h] + encB[h] ----------------
__global__ void k_encoder(const float* __restrict__ x, const float* __restrict__ W,
                          const float* __restrict__ bv, float* __restrict__ u) {
    int blk = blockIdx.x;
    int lt = blk % (LQ / 64);
    int b  = blk / (LQ / 64);
    int l0 = lt * 64;
    int t  = threadIdx.x;
    __shared__ float xs[64][DIN + 1];
    for (int idx = t; idx < 64 * DIN; idx += 256) {
        int ll = idx / DIN, d = idx % DIN;
        xs[ll][d] = x[((size_t)b * LQ + l0 + ll) * DIN + d];
    }
    __syncthreads();
    int ll = t & 63;
    int hb = t >> 6;                       // 0..3
    for (int hh = 0; hh < 64; ++hh) {
        int h = hb * 64 + hh;
        float acc = bv[h];
#pragma unroll
        for (int d = 0; d < DIN; ++d) acc = fmaf(xs[ll][d], W[d * H + h], acc);
        u[((size_t)b * H + h) * LQ + l0 + ll] = acc;
    }
}

// ---------------- per-layer modal params: w = exp(dt*A), ct = C*(w-1)/A ----------------
__global__ void k_modal(const float* __restrict__ logdt, const float* __restrict__ logAr,
                        const float* __restrict__ Aim,  const float* __restrict__ Cre,
                        const float* __restrict__ Cim,
                        float* __restrict__ wri, float* __restrict__ ctri) {
    int i = blockIdx.x * blockDim.x + threadIdx.x;   // h*NC + n
    if (i >= H * NC) return;
    int h = i / NC;
    float dt = expf(logdt[h]);
    float ar = -expf(logAr[i]);
    float ai = Aim[i];
    float dar = ar * dt, dai = ai * dt;
    float e  = expf(dar);
    float wr = e * cosf(dai), wi = e * sinf(dai);
    float nr = wr - 1.0f, ni = wi;
    float den = ar * ar + ai * ai;
    float qr = (nr * ar + ni * ai) / den;
    float qi = (ni * ar - nr * ai) / den;
    float cr = Cre[i], ci = Cim[i];
    ctri[2 * i]     = cr * qr - ci * qi;
    ctri[2 * i + 1] = cr * qi + ci * qr;
    wri[2 * i] = wr; wri[2 * i + 1] = wi;
}

// ---------------- chunk tables: K[h,j], M[h,j,n]=2*ct*w^{j+1}, P[h,t,n]=w^{C-1-t}, wc=w^C
__global__ void k_tables(const float* __restrict__ wri, const float* __restrict__ ctri,
                         float* __restrict__ Kt, float* __restrict__ Mt,
                         float* __restrict__ Pt, float* __restrict__ wc) {
    int h = blockIdx.x;
    int j = threadIdx.x;                 // 0..CHUNK-1
    float kacc = 0.f;
    for (int n = 0; n < NC; ++n) {
        int i = h * NC + n;
        float wr = wri[2 * i], wi = wri[2 * i + 1];
        float cr = ctri[2 * i], ci = ctri[2 * i + 1];
        float pr = 1.f, pi = 0.f;
        for (int s = 0; s < j; ++s) { float tr = pr * wr - pi * wi; pi = pr * wi + pi * wr; pr = tr; }
        kacc += 2.f * (cr * pr - ci * pi);
        float qr = pr * wr - pi * wi;                     // w^{j+1}
        float qi = pr * wi + pi * wr;
        size_t mi = ((size_t)(h * CHUNK + j) * NC + n) * 2;
        Mt[mi]     = 2.f * (cr * qr - ci * qi);
        Mt[mi + 1] = 2.f * (cr * qi + ci * qr);
        size_t pidx = ((size_t)(h * CHUNK + (CHUNK - 1 - j)) * NC + n) * 2;
        Pt[pidx] = pr; Pt[pidx + 1] = pi;
        if (j == CHUNK - 1) { wc[2 * i] = qr; wc[2 * i + 1] = qi; }
    }
    Kt[h * CHUNK + j] = kacc;
}

// ---------------- local chunk states: Sloc[b,ch,h,n] = sum_t P[h,t,n]*z[b,h,ch*C+t] ----
__global__ void k_sloc(const float* __restrict__ u, const float* __restrict__ Pt,
                       float* __restrict__ St) {
    int blk = blockIdx.x;
    int hg = blk % (H / 8);
    int ch = (blk / (H / 8)) % NCH;
    int b  = blk / ((H / 8) * NCH);
    int t  = threadIdx.x;
    int hs = t / NC;          // 0..7
    int n  = t % NC;
    int h  = hg * 8 + hs;
    __shared__ float zs[8][CHUNK];
    for (int idx = t; idx < 8 * CHUNK; idx += 256) {
        int hh = idx / CHUNK, ll = idx % CHUNK;
        zs[hh][ll] = u[((size_t)b * H + hg * 8 + hh) * LQ + ch * CHUNK + ll];
    }
    __syncthreads();
    float sr = 0.f, si = 0.f;
    const float* P = Pt + (size_t)h * CHUNK * NC * 2;
    for (int tt = 0; tt < CHUNK; ++tt) {
        float z = zs[hs][tt];
        sr = fmaf(P[(tt * NC + n) * 2],     z, sr);
        si = fmaf(P[(tt * NC + n) * 2 + 1], z, si);
    }
    size_t o = (((size_t)b * NCH + ch) * H + h) * NC + n;
    St[2 * o] = sr; St[2 * o + 1] = si;
}

// ---------------- sequential chunk scan (in place): St[ch] becomes carry-in S_in --------
__global__ void k_scan(const float* __restrict__ wc, float* __restrict__ St) {
    int i = blockIdx.x * blockDim.x + threadIdx.x;   // b*H*NC + hn
    if (i >= B * H * NC) return;
    int hn = i % (H * NC);
    int b  = i / (H * NC);
    float wr = wc[2 * hn], wi = wc[2 * hn + 1];
    float cr = 0.f, ci = 0.f;
    for (int ch = 0; ch < NCH; ++ch) {
        size_t o = 2 * (((size_t)b * NCH + ch) * H * NC + hn);
        float tr = St[o], ti = St[o + 1];
        St[o] = cr; St[o + 1] = ci;
        float nr = wr * cr - wi * ci + tr;
        ci = wr * ci + wi * cr + ti;
        cr = nr;
    }
}

// ---- y = local_triangular_conv + carry + D*z ; GELU(exact) ; write bf16 (B,L,H) -------
__global__ void k_conv(const float* __restrict__ u, const float* __restrict__ Kt,
                       const float* __restrict__ Mt, const float* __restrict__ St,
                       const float* __restrict__ Dp, unsigned short* __restrict__ ybf) {
    int blk = blockIdx.x;
    int hg = blk % (H / 64);
    int ch = (blk / (H / 64)) % NCH;
    int b  = blk / ((H / 64) * NCH);
    int t  = threadIdx.x;
    __shared__ float zs[64][65];
    __shared__ float Ks[64][65];
    __shared__ float Ss[64][64];
    __shared__ float Ds[64];
    for (int idx = t; idx < 64 * 64; idx += 256) {
        int hh = idx >> 6, ll = idx & 63;
        zs[hh][ll] = u[((size_t)b * H + hg * 64 + hh) * LQ + ch * CHUNK + ll];
        Ks[hh][ll] = Kt[(hg * 64 + hh) * CHUNK + ll];
        Ss[hh][ll] = St[(((size_t)b * NCH + ch) * H + hg * 64 + hh) * NC * 2 + ll];
    }
    if (t < 64) Ds[t] = Dp[hg * 64 + t];
    __syncthreads();
    int hl = t & 63;
    int lb = t >> 6;                     // 0..3 (wave-uniform)
    int h  = hg * 64 + hl;
    for (int li = 0; li < 16; ++li) {
        int l = lb + 4 * li;
        float acc = 0.f;
        for (int tt = 0; tt <= l; ++tt) acc = fmaf(Ks[hl][l - tt], zs[hl][tt], acc);
        const float* M = Mt + (size_t)(h * CHUNK + l) * NC * 2;
        float carry = 0.f;
#pragma unroll
        for (int n = 0; n < NC; ++n)
            carry += M[2 * n] * Ss[hl][2 * n] - M[2 * n + 1] * Ss[hl][2 * n + 1];
        float y = acc + carry + Ds[hl] * zs[hl][l];
        float g = 0.5f * y * (1.0f + erff(y * 0.70710678118654752f));
        ybf[((size_t)b * LQ + ch * CHUNK + l) * H + h] = f2bf(g);
    }
}

// ---------------- convert out_W layer slice to bf16 -----------------------------------
__global__ void k_cvtW(const float* __restrict__ W, unsigned short* __restrict__ Wbf) {
    int i = blockIdx.x * blockDim.x + threadIdx.x;
    if (i < 2 * H * H) Wbf[i] = f2bf(W[i]);
}

// ---- WMMA GEMM (M=L,N=2H paired as a/g,K=H) + bias + GLU + residual into u -----------
__global__ void __launch_bounds__(256) k_gemm_glu(const unsigned short* __restrict__ ybf,
                                                  const unsigned short* __restrict__ Wbf,
                                                  const float* __restrict__ ob,
                                                  float* __restrict__ u) {
    int wave = blockIdx.x * 8 + (threadIdx.x >> 5);
    int lane = threadIdx.x & 31;
    int o_t = wave & 15;                // H/16
    int l_t = (wave >> 4) & 127;        // LQ/32
    int b   = wave >> 11;
    int o0 = o_t * 16;
    int l0 = l_t * 32;
    int sel = lane >> 4;
    int mr  = lane & 15;

    v8f cA0 = {}, cA1 = {}, cG0 = {}, cG1 = {};
    union Frag { uint4 q[2]; v16bf v; };

    for (int k0 = 0; k0 < H; k0 += 32) {
        // A tiles: rows l0+mr and l0+16+mr, K pattern {0..7,16..23}/{8..15,24..31}
        const uint4* a0 = reinterpret_cast<const uint4*>(
            ybf + ((size_t)b * LQ + l0 + mr) * H + k0 + 8 * sel);
        const uint4* a1 = reinterpret_cast<const uint4*>(
            ybf + ((size_t)b * LQ + l0 + 16 + mr) * H + k0 + 8 * sel);
        Frag fa0, fa1;
        fa0.q[0] = a0[0]; fa0.q[1] = a0[2];    // +16 halves = +2 uint4
        fa1.q[0] = a1[0]; fa1.q[1] = a1[2];
        // B tiles: col o0+mr (a) and H+o0+mr (g); K pattern {0..7,16..23}/{8..15,24..31}
        const uint4* bwa = reinterpret_cast<const uint4*>(
            Wbf + (size_t)(o0 + mr) * H + k0 + 8 * sel);
        const uint4* bwg = reinterpret_cast<const uint4*>(
            Wbf + (size_t)(H + o0 + mr) * H + k0 + 8 * sel);
        Frag fba, fbg;
        fba.q[0] = bwa[0]; fba.q[1] = bwa[2];
        fbg.q[0] = bwg[0]; fbg.q[1] = bwg[2];

        cA0 = __builtin_amdgcn_wmma_f32_16x16x32_bf16(false, fa0.v, false, fba.v,
                                                      (short)0, cA0, false, false);
        cA1 = __builtin_amdgcn_wmma_f32_16x16x32_bf16(false, fa1.v, false, fba.v,
                                                      (short)0, cA1, false, false);
        cG0 = __builtin_amdgcn_wmma_f32_16x16x32_bf16(false, fa0.v, false, fbg.v,
                                                      (short)0, cG0, false, false);
        cG1 = __builtin_amdgcn_wmma_f32_16x16x32_bf16(false, fa1.v, false, fbg.v,
                                                      (short)0, cG1, false, false);
    }

    int o = o0 + mr;                    // D-matrix: N = lane%16
    float ba = ob[o], bg = ob[H + o];
#pragma unroll
    for (int g = 0; g < 8; ++g) {
        int l = l0 + g + 8 * sel;       // D-matrix: M = g + 8*(lane/16)
        {
            float a  = cA0[g] + ba;
            float gg = cG0[g] + bg;
            float z  = a * (1.0f / (1.0f + expf(-gg)));
            size_t idx = ((size_t)b * H + o) * LQ + l;
            u[idx] = z + u[idx];
        }
        {
            float a  = cA1[g] + ba;
            float gg = cG1[g] + bg;
            float z  = a * (1.0f / (1.0f + expf(-gg)));
            size_t idx = ((size_t)b * H + o) * LQ + (l + 16);
            u[idx] = z + u[idx];
        }
    }
}

// ---------------- LayerNorm over channel dim (in place) --------------------------------
__global__ void k_ln(float* __restrict__ u, const float* __restrict__ g,
                     const float* __restrict__ bb) {
    int i = blockIdx.x * blockDim.x + threadIdx.x;   // b*LQ + l
    int b = i / LQ, l = i % LQ;
    float s = 0.f, s2 = 0.f;
    for (int h = 0; h < H; ++h) {
        float v = u[((size_t)b * H + h) * LQ + l];
        s += v; s2 += v * v;
    }
    float mu  = s * (1.0f / H);
    float var = s2 * (1.0f / H) - mu * mu;
    float r   = rsqrtf(var + LN_EPS);
    for (int h = 0; h < H; ++h) {
        size_t idx = ((size_t)b * H + h) * LQ + l;
        u[idx] = (u[idx] - mu) * r * g[h] + bb[h];
    }
}

// ---------------- mean pool over L -----------------------------------------------------
__global__ void k_pool(const float* __restrict__ u, float* __restrict__ pooled) {
    int wave = blockIdx.x * 8 + (threadIdx.x >> 5);   // b*H waves
    int lane = threadIdx.x & 31;
    int b = wave / H, h = wave % H;
    float s = 0.f;
    for (int l = lane; l < LQ; l += 32) s += u[((size_t)b * H + h) * LQ + l];
    for (int off = 16; off; off >>= 1) s += __shfl_down(s, off);
    if (lane == 0) pooled[b * H + h] = s * (1.0f / LQ);
}

// ---------------- decoder --------------------------------------------------------------
__global__ void k_dec(const float* __restrict__ pooled, const float* __restrict__ dW,
                      const float* __restrict__ db, float* __restrict__ out) {
    int i = blockIdx.x * blockDim.x + threadIdx.x;
    if (i >= B * DOUT) return;
    int b = i / DOUT, d = i % DOUT;
    float s = db[d];
    for (int h = 0; h < H; ++h) s = fmaf(pooled[b * H + h], dW[h * DOUT + d], s);
    out[i] = s;
}

extern "C" void kernel_launch(void* const* d_in, const int* in_sizes, int n_in,
                              void* d_out, int out_size, void* d_ws, size_t ws_size,
                              hipStream_t stream) {
    const float* x     = (const float*)d_in[0];
    const float* encW  = (const float*)d_in[1];
    const float* encB  = (const float*)d_in[2];
    const float* logdt = (const float*)d_in[3];
    const float* logAr = (const float*)d_in[4];
    const float* Aim   = (const float*)d_in[5];
    const float* Cre   = (const float*)d_in[6];
    const float* Cim   = (const float*)d_in[7];
    const float* Dp    = (const float*)d_in[8];
    const float* outW  = (const float*)d_in[9];
    const float* outb  = (const float*)d_in[10];
    const float* lng   = (const float*)d_in[11];
    const float* lnb   = (const float*)d_in[12];
    const float* decW  = (const float*)d_in[13];
    const float* decb  = (const float*)d_in[14];
    float* out = (float*)d_out;

    char* ws = (char*)d_ws;
    size_t off = 0;
    auto alloc = [&](size_t bytes) -> char* {
        char* p = ws + off;
        off += (bytes + 255) & ~(size_t)255;
        return p;
    };
    float*          u    = (float*)alloc((size_t)B * H * LQ * 4);           // 67 MB
    unsigned short* ybf  = (unsigned short*)alloc((size_t)B * LQ * H * 2);  // 33 MB
    unsigned short* Wbf  = (unsigned short*)alloc((size_t)2 * H * H * 2);
    float*          wri  = (float*)alloc((size_t)H * NC * 2 * 4);
    float*          ctri = (float*)alloc((size_t)H * NC * 2 * 4);
    float*          wc   = (float*)alloc((size_t)H * NC * 2 * 4);
    float*          Kt   = (float*)alloc((size_t)H * CHUNK * 4);
    float*          Mt   = (float*)alloc((size_t)H * CHUNK * NC * 2 * 4);
    float*          Pt   = (float*)alloc((size_t)H * CHUNK * NC * 2 * 4);
    float*          St   = (float*)alloc((size_t)B * NCH * H * NC * 2 * 4); // 67 MB
    float*          pool = (float*)alloc((size_t)B * H * 4);

    k_encoder<<<B * (LQ / 64), 256, 0, stream>>>(x, encW, encB, u);

    for (int i = 0; i < NL; ++i) {
        k_modal<<<(H * NC + 255) / 256, 256, 0, stream>>>(
            logdt + (size_t)i * H, logAr + (size_t)i * H * NC, Aim + (size_t)i * H * NC,
            Cre + (size_t)i * H * NC, Cim + (size_t)i * H * NC, wri, ctri);
        k_tables<<<H, CHUNK, 0, stream>>>(wri, ctri, Kt, Mt, Pt, wc);
        k_sloc<<<B * NCH * (H / 8), 256, 0, stream>>>(u, Pt, St);
        k_scan<<<(B * H * NC + 255) / 256, 256, 0, stream>>>(wc, St);
        k_conv<<<B * NCH * (H / 64), 256, 0, stream>>>(u, Kt, Mt, St, Dp + (size_t)i * H, ybf);
        k_cvtW<<<(2 * H * H + 255) / 256, 256, 0, stream>>>(outW + (size_t)i * 2 * H * H, Wbf);
        k_gemm_glu<<<(B * (LQ / 32) * (H / 16)) / 8, 256, 0, stream>>>(
            ybf, Wbf, outb + (size_t)i * 2 * H, u);
        k_ln<<<(B * LQ) / 256, 256, 0, stream>>>(u, lng + (size_t)i * H, lnb + (size_t)i * H);
    }

    k_pool<<<(B * H) / 8, 256, 0, stream>>>(u, pool);
    k_dec<<<1, 256, 0, stream>>>(pool, decW, decb, out);
}
